// SwinTransformerBlock_87058987090295
// MI455X (gfx1250) — compile-verified, matrix operand order checked by reading, not run on hardware
//
#include <hip/hip_runtime.h>
#include <math.h>
#include <stdint.h>

typedef __attribute__((ext_vector_type(16))) _Float16 v16h;
typedef __attribute__((ext_vector_type(8)))  _Float16 v8h;
typedef __attribute__((ext_vector_type(8)))  float    v8f;

// Problem constants
#define TOK   50176      // B * H * W tokens (64*784), = 128*392
#define CD    384
#define QKVD  1152
#define HIDD  1536
#define NHEAD 12
#define NWIN  1024       // B * 16 windows
#define WTOK  49
#define SCALE_ATTN 0.17677669529663687f  // 32^-0.5

#define USE_ASYNC_LDS 1

// Async global->LDS 16B copy (GVS mode: saddr base + 32-bit vgpr byte offset).
__device__ __forceinline__ void async_cp16(const void* base, unsigned byte_off,
                                           unsigned lds_off) {
#if USE_ASYNC_LDS
  asm volatile("global_load_async_to_lds_b128 %0, %1, %2"
               :: "v"(lds_off), "v"(byte_off), "s"(base) : "memory");
#endif
}

// ---------------------------------------------------------------------------
// fp32 -> fp16 convert (weights)
// ---------------------------------------------------------------------------
__global__ __launch_bounds__(256) void cvt_f32_f16(const float* __restrict__ s,
                                                   _Float16* __restrict__ d, int n) {
  int i = blockIdx.x * 256 + threadIdx.x;
  if (i < n) d[i] = (_Float16)s[i];
}

// ---------------------------------------------------------------------------
// Precompute combined relpos-bias + shift-mask:  bm[wloc][head][64][64] f32
// (3 MB total -> L2 resident; removes per-element index math from attention)
// ---------------------------------------------------------------------------
__global__ __launch_bounds__(256) void build_biasmask(const float* __restrict__ btab,
                                                      float* __restrict__ bm) {
  const int wloc = blockIdx.x / NHEAD;
  const int head = blockIdx.x % NHEAD;
  const int wh = wloc >> 2, ww = wloc & 3;
  float* dst = bm + (size_t)blockIdx.x * 4096;
  for (int e = threadIdx.x; e < 4096; e += 256) {
    const int n = e >> 6, m = e & 63;
    float val;
    if (m >= WTOK) {
      val = -1e30f;
    } else {
      const int nn = (n < WTOK) ? n : (WTOK - 1);
      const int in_ = nn / 7, jn = nn % 7, im = m / 7, jm = m % 7;
      const int ridx = (in_ - im + 6) * 13 + (jn - jm + 6);
      const float bv = btab[ridx * NHEAD + head];
      const int ghn = wh * 7 + in_, gwn = ww * 7 + jn;
      const int ghm = wh * 7 + im,  gwm = ww * 7 + jm;
      const int rn = (ghn < 21 ? 0 : (ghn < 25 ? 1 : 2)) * 3 +
                     (gwn < 21 ? 0 : (gwn < 25 ? 1 : 2));
      const int rm = (ghm < 21 ? 0 : (ghm < 25 ? 1 : 2)) * 3 +
                     (gwm < 21 ? 0 : (gwm < 25 ? 1 : 2));
      val = bv + (rn != rm ? -100.0f : 0.0f);
    }
    dst[e] = val;
  }
}

// ---------------------------------------------------------------------------
// LayerNorm1 + roll(-3,-3) + window partition.  One wave per output token row.
// ---------------------------------------------------------------------------
__global__ __launch_bounds__(256) void ln1_partition(const float* __restrict__ x,
                                                     const float* __restrict__ w,
                                                     const float* __restrict__ b,
                                                     _Float16* __restrict__ out) {
  const int wave = threadIdx.x >> 5, lane = threadIdx.x & 31;
  const int r = blockIdx.x * 8 + wave;
  const int bimg = r / 784;
  const int rem  = r % 784;
  const int wl   = rem / WTOK;
  const int n    = rem % WTOK;
  const int hp   = (wl >> 2) * 7 + n / 7;
  const int wp   = (wl & 3) * 7 + n % 7;
  const int hs   = (hp + 3) % 28;
  const int ws_  = (wp + 3) % 28;
  const float* src = x + ((long)bimg * 784 + hs * 28 + ws_) * CD;

  float v[12], s = 0.f;
#pragma unroll
  for (int k = 0; k < 12; k++) { v[k] = src[lane + 32 * k]; s += v[k]; }
#pragma unroll
  for (int off = 16; off >= 1; off >>= 1) s += __shfl_xor(s, off, 32);
  const float mean = s * (1.0f / CD);
  float s2 = 0.f;
#pragma unroll
  for (int k = 0; k < 12; k++) { float d = v[k] - mean; s2 += d * d; }
#pragma unroll
  for (int off = 16; off >= 1; off >>= 1) s2 += __shfl_xor(s2, off, 32);
  const float rstd = rsqrtf(s2 * (1.0f / CD) + 1e-5f);
  _Float16* dst = out + (long)r * CD;
#pragma unroll
  for (int k = 0; k < 12; k++) {
    int c = lane + 32 * k;
    dst[c] = (_Float16)((v[k] - mean) * rstd * w[c] + b[c]);
  }
}

// ---------------------------------------------------------------------------
// Plain LayerNorm (token-major), f32 in -> f16 out
// ---------------------------------------------------------------------------
__global__ __launch_bounds__(256) void ln2_kernel(const float* __restrict__ x,
                                                  const float* __restrict__ w,
                                                  const float* __restrict__ b,
                                                  _Float16* __restrict__ out) {
  const int wave = threadIdx.x >> 5, lane = threadIdx.x & 31;
  const int r = blockIdx.x * 8 + wave;
  const float* src = x + (long)r * CD;
  float v[12], s = 0.f;
#pragma unroll
  for (int k = 0; k < 12; k++) { v[k] = src[lane + 32 * k]; s += v[k]; }
#pragma unroll
  for (int off = 16; off >= 1; off >>= 1) s += __shfl_xor(s, off, 32);
  const float mean = s * (1.0f / CD);
  float s2 = 0.f;
#pragma unroll
  for (int k = 0; k < 12; k++) { float d = v[k] - mean; s2 += d * d; }
#pragma unroll
  for (int off = 16; off >= 1; off >>= 1) s2 += __shfl_xor(s2, off, 32);
  const float rstd = rsqrtf(s2 * (1.0f / CD) + 1e-5f);
  _Float16* dst = out + (long)r * CD;
#pragma unroll
  for (int k = 0; k < 12; k++) {
    int c = lane + 32 * k;
    dst[c] = (_Float16)((v[k] - mean) * rstd * w[c] + b[c]);
  }
}

// ---------------------------------------------------------------------------
// Window reverse + roll(+3,+3) + residual:  x2 = x + gather(projOut)
// ---------------------------------------------------------------------------
__global__ __launch_bounds__(256) void add_unpartition(const float* __restrict__ x,
                                                       const float* __restrict__ po,
                                                       float* __restrict__ x2) {
  const int idx4 = blockIdx.x * 256 + threadIdx.x;
  const int idx  = idx4 * 4;
  const int row  = idx / CD;
  const int c    = idx % CD;
  const int bimg = row / 784, rem = row % 784;
  const int h = rem / 28, w = rem % 28;
  const int hp = (h + 25) % 28;
  const int wp = (w + 25) % 28;
  const int wrow = bimg * 784 + ((hp / 7) * 4 + wp / 7) * WTOK + (hp % 7) * 7 + (wp % 7);
  float4 a = *(const float4*)(x + idx);
  float4 p = *(const float4*)(po + (long)wrow * CD + c);
  float4 o; o.x = a.x + p.x; o.y = a.y + p.y; o.z = a.z + p.z; o.w = a.w + p.w;
  *(float4*)(x2 + idx) = o;
}

// ---------------------------------------------------------------------------
// Tiled WMMA GEMM: C[M,N] = A[M,K] * Bt[N,K]^T + bias.
// Block tile 128x128, BK=32, double-buffered LDS fed by async global->LDS.
// 8 waves (4x2), each wave 32x64 = 2x4 WMMA 16x16x32 f16 tiles.
// EP: 0 = store f16, 1 = store f32, 2 = GELU(exact) -> f16, 3 = +res -> f32
// ---------------------------------------------------------------------------
template <int EP>
__global__ __launch_bounds__(256) void gemm_wmma(const _Float16* __restrict__ A,
                                                 const _Float16* __restrict__ Bt,
                                                 const float* __restrict__ bias,
                                                 int K, int N,
                                                 _Float16* __restrict__ out16,
                                                 float* __restrict__ out32,
                                                 const float* __restrict__ res) {
  constexpr int BM = 128, BN = 128, BK = 32, LDP = 40;
  __shared__ __align__(16) _Float16 sA[2][BM * LDP];
  __shared__ __align__(16) _Float16 sB[2][BN * LDP];

  const int tid  = threadIdx.x;
  const int wave = tid >> 5, lane = tid & 31;
  const int wm = wave >> 1, wn = wave & 1;
  const int mBase = blockIdx.y * BM;
  const int nBase = blockIdx.x * BN;
  const int halfsel = lane >> 4;
  const int l15 = lane & 15;
  const int kb = halfsel * 8;

  // Per-thread staging coordinates (4 x 16B chunks: 2 for A, 2 for B)
  const int rA0 = (tid * 2) >> 2,     cA0 = (tid * 2) & 3;
  const int rA1 = (tid * 2 + 1) >> 2, cA1 = (tid * 2 + 1) & 3;

  v8f acc[2][4];
#pragma unroll
  for (int i = 0; i < 2; i++)
#pragma unroll
    for (int j = 0; j < 4; j++) acc[i][j] = (v8f){};

  auto stage = [&](int k0, int buf) {
#if USE_ASYNC_LDS
    async_cp16(A, (unsigned)(((mBase + rA0) * K + k0 + cA0 * 8) * 2),
               (unsigned)(uintptr_t)&sA[buf][rA0 * LDP + cA0 * 8]);
    async_cp16(A, (unsigned)(((mBase + rA1) * K + k0 + cA1 * 8) * 2),
               (unsigned)(uintptr_t)&sA[buf][rA1 * LDP + cA1 * 8]);
    async_cp16(Bt, (unsigned)(((nBase + rA0) * K + k0 + cA0 * 8) * 2),
               (unsigned)(uintptr_t)&sB[buf][rA0 * LDP + cA0 * 8]);
    async_cp16(Bt, (unsigned)(((nBase + rA1) * K + k0 + cA1 * 8) * 2),
               (unsigned)(uintptr_t)&sB[buf][rA1 * LDP + cA1 * 8]);
#else
    *(uint4*)(&sA[buf][rA0 * LDP + cA0 * 8]) =
        *(const uint4*)(A + (mBase + rA0) * K + k0 + cA0 * 8);
    *(uint4*)(&sA[buf][rA1 * LDP + cA1 * 8]) =
        *(const uint4*)(A + (mBase + rA1) * K + k0 + cA1 * 8);
    *(uint4*)(&sB[buf][rA0 * LDP + cA0 * 8]) =
        *(const uint4*)(Bt + (nBase + rA0) * K + k0 + cA0 * 8);
    *(uint4*)(&sB[buf][rA1 * LDP + cA1 * 8]) =
        *(const uint4*)(Bt + (nBase + rA1) * K + k0 + cA1 * 8);
#endif
  };

  const int nk = K / BK;
  stage(0, 0);

  for (int ks = 0; ks < nk; ks++) {
    const int buf = ks & 1;
    if (ks + 1 < nk) {
      stage((ks + 1) * BK, buf ^ 1);
#if USE_ASYNC_LDS
      asm volatile("s_wait_asynccnt 0x4" ::: "memory");
#endif
    } else {
#if USE_ASYNC_LDS
      asm volatile("s_wait_asynccnt 0x0" ::: "memory");
#endif
    }
    __syncthreads();

    v16h af[2], bf[4];
#pragma unroll
    for (int i = 0; i < 2; i++) {
      const int r = wm * 32 + i * 16 + l15;
      v8h lo = *(const v8h*)(&sA[buf][r * LDP + kb]);
      v8h hi = *(const v8h*)(&sA[buf][r * LDP + kb + 16]);
#pragma unroll
      for (int t = 0; t < 8; t++) { af[i][t] = lo[t]; af[i][8 + t] = hi[t]; }
    }
#pragma unroll
    for (int j = 0; j < 4; j++) {
      const int r = wn * 64 + j * 16 + l15;
      v8h lo = *(const v8h*)(&sB[buf][r * LDP + kb]);
      v8h hi = *(const v8h*)(&sB[buf][r * LDP + kb + 16]);
#pragma unroll
      for (int t = 0; t < 8; t++) { bf[j][t] = lo[t]; bf[j][8 + t] = hi[t]; }
    }
#pragma unroll
    for (int i = 0; i < 2; i++)
#pragma unroll
      for (int j = 0; j < 4; j++)
        acc[i][j] = __builtin_amdgcn_wmma_f32_16x16x32_f16(
            false, af[i], false, bf[j], (short)0, acc[i][j], false, false);
    __syncthreads();
  }

  // Epilogue
#pragma unroll
  for (int i = 0; i < 2; i++) {
#pragma unroll
    for (int j = 0; j < 4; j++) {
      const int gcol = nBase + wn * 64 + j * 16 + l15;
      const float bv = bias[gcol];
#pragma unroll
      for (int e = 0; e < 8; e++) {
        const int grow = mBase + wm * 32 + i * 16 + halfsel * 8 + e;
        float v = acc[i][j][e] + bv;
        if (EP == 2) v = 0.5f * v * (1.0f + erff(v * 0.70710678118654752f));
        if (EP == 3) v += res[grow * N + gcol];
        if (EP == 0 || EP == 2) out16[grow * N + gcol] = (_Float16)v;
        else                    out32[grow * N + gcol] = v;
      }
    }
  }
}

// ---------------------------------------------------------------------------
// Attention: one wave per (window, head).  49 tokens padded to 64.
// S = Q K^T * scale + bm[wloc][head]; softmax; O = P V.
// ---------------------------------------------------------------------------
__global__ __launch_bounds__(32) void attn_kernel(const _Float16* __restrict__ qkv,
                                                  const float* __restrict__ bm,
                                                  _Float16* __restrict__ oBuf) {
  constexpr int LDQ = 40, LDV = 72, LDS_ = 65, LPP = 72;
  __shared__ __align__(16) _Float16 Qs[64 * LDQ];
  __shared__ __align__(16) _Float16 Ks[64 * LDQ];
  __shared__ __align__(16) _Float16 Vt[32 * LDV];
  __shared__ __align__(16) float    Sf[64 * LDS_];
  __shared__ __align__(16) _Float16 Pp[64 * LPP];

  const int lane = threadIdx.x;
  const int win  = blockIdx.x / NHEAD;
  const int head = blockIdx.x % NHEAD;
  const int wloc = win & 15;
  const long baseQ = (long)win * WTOK * QKVD + head * 32;
  const float* bmw = bm + ((size_t)(wloc * NHEAD + head)) * 4096;

  for (int idx = lane; idx < 256; idx += 32) {
    const int row = idx >> 2, c8 = idx & 3;
    uint4 zq = {0, 0, 0, 0}, zk = {0, 0, 0, 0};
    v8h zv = {};
    if (row < WTOK) {
      zq = *(const uint4*)(qkv + baseQ + row * QKVD + c8 * 8);
      zk = *(const uint4*)(qkv + baseQ + CD + row * QKVD + c8 * 8);
      zv = *(const v8h*)(qkv + baseQ + 2 * CD + row * QKVD + c8 * 8);
    }
    *(uint4*)(&Qs[row * LDQ + c8 * 8]) = zq;
    *(uint4*)(&Ks[row * LDQ + c8 * 8]) = zk;
#pragma unroll
    for (int u = 0; u < 8; u++) Vt[(c8 * 8 + u) * LDV + row] = zv[u];
  }
  __syncthreads();

  const int halfsel = lane >> 4, l15 = lane & 15, kbq = halfsel * 8;

  // S = Q K^T  (4x4 tiles, single K=32 step each)
#pragma unroll
  for (int ti = 0; ti < 4; ti++) {
    v16h aq;
    {
      const int r = ti * 16 + l15;
      v8h lo = *(const v8h*)(&Qs[r * LDQ + kbq]);
      v8h hi = *(const v8h*)(&Qs[r * LDQ + kbq + 16]);
#pragma unroll
      for (int t = 0; t < 8; t++) { aq[t] = lo[t]; aq[8 + t] = hi[t]; }
    }
#pragma unroll
    for (int tj = 0; tj < 4; tj++) {
      v16h bk;
      {
        const int r = tj * 16 + l15;
        v8h lo = *(const v8h*)(&Ks[r * LDQ + kbq]);
        v8h hi = *(const v8h*)(&Ks[r * LDQ + kbq + 16]);
#pragma unroll
        for (int t = 0; t < 8; t++) { bk[t] = lo[t]; bk[8 + t] = hi[t]; }
      }
      v8f s = {};
      s = __builtin_amdgcn_wmma_f32_16x16x32_f16(false, aq, false, bk,
                                                 (short)0, s, false, false);
#pragma unroll
      for (int e = 0; e < 8; e++) {
        const int n = ti * 16 + halfsel * 8 + e;
        const int m = tj * 16 + l15;
        Sf[n * LDS_ + m] = s[e] * SCALE_ATTN + bmw[n * 64 + m];
      }
    }
  }
  __syncthreads();

  // Softmax: each lane owns rows {lane, lane+32}; stride-65 rows conflict-free
#pragma unroll
  for (int rr = 0; rr < 2; rr++) {
    const int r = lane + rr * 32;
    float mx = -1e30f;
    for (int c = 0; c < 64; c++) mx = fmaxf(mx, Sf[r * LDS_ + c]);
    float sum = 0.f;
    for (int c = 0; c < 64; c++) {
      float e = __expf(Sf[r * LDS_ + c] - mx);
      Sf[r * LDS_ + c] = e;
      sum += e;
    }
    const float inv = 1.0f / sum;
    for (int c = 0; c < 64; c++) Pp[r * LPP + c] = (_Float16)(Sf[r * LDS_ + c] * inv);
  }
  __syncthreads();

  // O = P V : 4x2 tiles, K=64 (two k-steps)
  const long baseO = (long)win * WTOK * CD + head * 32;
#pragma unroll
  for (int ti = 0; ti < 4; ti++) {
    v8f o0 = {}, o1 = {};
#pragma unroll
    for (int kk = 0; kk < 2; kk++) {
      v16h ap;
      {
        const int r = ti * 16 + l15;
        const int kofs = kk * 32 + kbq;
        v8h lo = *(const v8h*)(&Pp[r * LPP + kofs]);
        v8h hi = *(const v8h*)(&Pp[r * LPP + kofs + 16]);
#pragma unroll
        for (int t = 0; t < 8; t++) { ap[t] = lo[t]; ap[8 + t] = hi[t]; }
      }
      v16h bv0, bv1;
      {
        const int kofs = kk * 32 + kbq;
        const int r0 = l15, r1 = 16 + l15;
        v8h lo0 = *(const v8h*)(&Vt[r0 * LDV + kofs]);
        v8h hi0 = *(const v8h*)(&Vt[r0 * LDV + kofs + 16]);
        v8h lo1 = *(const v8h*)(&Vt[r1 * LDV + kofs]);
        v8h hi1 = *(const v8h*)(&Vt[r1 * LDV + kofs + 16]);
#pragma unroll
        for (int t = 0; t < 8; t++) {
          bv0[t] = lo0[t]; bv0[8 + t] = hi0[t];
          bv1[t] = lo1[t]; bv1[8 + t] = hi1[t];
        }
      }
      o0 = __builtin_amdgcn_wmma_f32_16x16x32_f16(false, ap, false, bv0,
                                                  (short)0, o0, false, false);
      o1 = __builtin_amdgcn_wmma_f32_16x16x32_f16(false, ap, false, bv1,
                                                  (short)0, o1, false, false);
    }
#pragma unroll
    for (int e = 0; e < 8; e++) {
      const int n = ti * 16 + halfsel * 8 + e;
      if (n < WTOK) {
        oBuf[baseO + n * CD + l15]      = (_Float16)o0[e];
        oBuf[baseO + n * CD + 16 + l15] = (_Float16)o1[e];
      }
    }
  }
}

// ---------------------------------------------------------------------------
// Launch
// ---------------------------------------------------------------------------
extern "C" void kernel_launch(void* const* d_in, const int* in_sizes, int n_in,
                              void* d_out, int out_size, void* d_ws, size_t ws_size,
                              hipStream_t stream) {
  const float* x       = (const float*)d_in[0];
  const float* n1w     = (const float*)d_in[2];
  const float* n1b     = (const float*)d_in[3];
  const float* qkv_w   = (const float*)d_in[4];
  const float* qkv_b   = (const float*)d_in[5];
  const float* btab    = (const float*)d_in[6];
  const float* proj_w  = (const float*)d_in[7];
  const float* proj_b  = (const float*)d_in[8];
  const float* n2w     = (const float*)d_in[9];
  const float* n2b     = (const float*)d_in[10];
  const float* fc1_w   = (const float*)d_in[11];
  const float* fc1_b   = (const float*)d_in[12];
  const float* fc2_w   = (const float*)d_in[13];
  const float* fc2_b   = (const float*)d_in[14];

  char* ws = (char*)d_ws;
  const size_t szA = (size_t)TOK * CD * 2;        // hWin / m16
  const size_t szQ = (size_t)TOK * QKVD * 2;      // qkv16
  const size_t szB = (size_t)TOK * HIDD * 2;      // fc1out (spans qkv+oBuf)
  const size_t szF = (size_t)TOK * CD * 4;        // f32 token tensor

  _Float16* hWin    = (_Float16*)(ws);
  _Float16* m16     = hWin;
  _Float16* qkv16   = (_Float16*)(ws + szA);
  _Float16* oBuf    = (_Float16*)(ws + szA + szQ);
  _Float16* fc1out  = qkv16;
  float*    x2      = (float*)(ws + szA + szB);
  float*    projOut = (float*)(ws + szA + szB + szF);
  char*     wsw     = ws + szA + szB + 2 * szF;
  _Float16* qkv_w16 = (_Float16*)(wsw);
  _Float16* proj_w16= (_Float16*)(wsw + (size_t)QKVD * CD * 2);
  _Float16* fc1_w16 = (_Float16*)(wsw + (size_t)QKVD * CD * 2 + (size_t)CD * CD * 2);
  _Float16* fc2_w16 = (_Float16*)(wsw + (size_t)QKVD * CD * 2 + (size_t)CD * CD * 2 +
                                  (size_t)HIDD * CD * 2);
  float*    bm      = (float*)(wsw + (size_t)QKVD * CD * 2 + (size_t)CD * CD * 2 +
                               (size_t)HIDD * CD * 2 + (size_t)CD * HIDD * 2);
  float* out = (float*)d_out;

  // 1) weight converts + bias/mask table
  cvt_f32_f16<<<(QKVD * CD + 255) / 256, 256, 0, stream>>>(qkv_w, qkv_w16, QKVD * CD);
  cvt_f32_f16<<<(CD * CD + 255) / 256, 256, 0, stream>>>(proj_w, proj_w16, CD * CD);
  cvt_f32_f16<<<(HIDD * CD + 255) / 256, 256, 0, stream>>>(fc1_w, fc1_w16, HIDD * CD);
  cvt_f32_f16<<<(CD * HIDD + 255) / 256, 256, 0, stream>>>(fc2_w, fc2_w16, CD * HIDD);
  build_biasmask<<<16 * NHEAD, 256, 0, stream>>>(btab, bm);

  // 2) LN1 + shift + partition
  ln1_partition<<<TOK / 8, 256, 0, stream>>>(x, n1w, n1b, hWin);

  // 3) QKV GEMM -> f16
  gemm_wmma<0><<<dim3(QKVD / 128, TOK / 128), 256, 0, stream>>>(
      hWin, qkv_w16, qkv_b, CD, QKVD, qkv16, nullptr, nullptr);

  // 4) windowed attention
  attn_kernel<<<NWIN * NHEAD, 32, 0, stream>>>(qkv16, bm, oBuf);

  // 5) proj GEMM -> f32
  gemm_wmma<1><<<dim3(CD / 128, TOK / 128), 256, 0, stream>>>(
      oBuf, proj_w16, proj_b, CD, CD, nullptr, projOut, nullptr);

  // 6) reverse + unshift + residual
  add_unpartition<<<(TOK * CD / 4) / 256, 256, 0, stream>>>(x, projOut, x2);

  // 7) LN2
  ln2_kernel<<<TOK / 8, 256, 0, stream>>>(x2, n2w, n2b, m16);

  // 8) FC1 GEMM + exact GELU -> f16
  gemm_wmma<2><<<dim3(HIDD / 128, TOK / 128), 256, 0, stream>>>(
      m16, fc1_w16, fc1_b, CD, HIDD, fc1out, nullptr, nullptr);

  // 9) FC2 GEMM + residual -> d_out (f32)
  gemm_wmma<3><<<dim3(CD / 128, TOK / 128), 256, 0, stream>>>(
      fc1out, fc2_w16, fc2_b, HIDD, CD, nullptr, out, x2);
}
